// SwinTransformerBlock3D_F_37649683316963
// MI455X (gfx1250) — matrix operand order repaired and checked by reading, not compiled, not run on hardware
//
#include <hip/hip_runtime.h>
#include <hip/hip_bf16.h>
#include <math.h>

typedef __attribute__((ext_vector_type(16))) _Float16 v16h;
typedef __attribute__((ext_vector_type(8)))  _Float16 v8h;
typedef __attribute__((ext_vector_type(8)))  float    v8f;

#define kN   98    // tokens per window
#define kNP  112   // padded rows (7 * 16)
#define kNH  4
#define kSCALE 0.17677669529663687f  /* 32^-0.5 */

union HV { v16h v; v8h h[2]; };

static __device__ inline v8f wmma_f16(v16h a, v16h b, v8f c) {
  // D = A(16x32 f16) * B(32x16 f16) + C(16x16 f32)
  return __builtin_amdgcn_wmma_f32_16x16x32_f16(false, a, false, b, (short)0, c, false, false);
}

static __device__ inline v8f vfill(float x) {
  v8f r;
#pragma unroll
  for (int i = 0; i < 8; ++i) r[i] = x;
  return r;
}

static __device__ inline float wave_sum(float v) {
#pragma unroll
  for (int o = 16; o > 0; o >>= 1) v += __shfl_xor(v, o, 32);
  return v;
}
static __device__ inline float wave_max(float v) {
#pragma unroll
  for (int o = 16; o > 0; o >>= 1) v = fmaxf(v, __shfl_xor(v, o, 32));
  return v;
}

// A fragment (16x32 f16) from row-major LDS (stride 128 halves).
// lanes 0-15: row=lane, K [0..15]; lanes 16-31: row=lane-16, K [16..31]
static __device__ inline v16h load_a(const _Float16* base, int row, int kbase, int lane) {
  const _Float16* p = base + row * 128 + kbase + ((lane >> 4) << 4);
  HV u;
  u.h[0] = *(const v8h*)(p);
  u.h[1] = *(const v8h*)(p + 8);
  return u.v;
}

// B fragment (32x16 f16) from column-major LDS (stride 128 halves per column).
// lane l: col = l&15, 16 contiguous K starting at kbase + (l>>4)*16
static __device__ inline v16h load_b(const _Float16* base, int col, int kbase, int lane) {
  const _Float16* p = base + col * 128 + kbase + ((lane >> 4) << 4);
  HV u;
  u.h[0] = *(const v8h*)(p);
  u.h[1] = *(const v8h*)(p + 8);
  return u.v;
}

// B fragment (32x16 f16) converted on the fly from a row-major fp32 weight
// matrix W[out,in] (in contiguous): column o of B == row o of W.
static __device__ inline v16h load_b_w32(const float* W, int ocol, int kbase, int lane) {
  const float* wp = W + (size_t)ocol * 128 + kbase + ((lane >> 4) << 4);
  float fl[16];
  *(float4*)(&fl[0])  = ((const float4*)wp)[0];
  *(float4*)(&fl[4])  = ((const float4*)wp)[1];
  *(float4*)(&fl[8])  = ((const float4*)wp)[2];
  *(float4*)(&fl[12]) = ((const float4*)wp)[3];
  v16h bf;
#pragma unroll
  for (int j = 0; j < 16; ++j) bf[j] = (_Float16)fl[j];
  return bf;
}

__global__ __launch_bounds__(256)
void swin3d_window_kernel(const float* __restrict__ x,
                          const float* __restrict__ mask,
                          const float* __restrict__ gamma,
                          const float* __restrict__ beta,
                          const float* __restrict__ wqkv,
                          const float* __restrict__ bqkv,
                          const float* __restrict__ wproj,
                          const float* __restrict__ bproj,
                          const float* __restrict__ btab,
                          const int*   __restrict__ rel,
                          float* __restrict__ out) {
  // 147.9 KB static LDS total -> 2 workgroups / 320KB WGP
  __shared__ __align__(16) _Float16 sXC[kNP * 128]; // LN(x); later ctx (attn@v)
  __shared__ __align__(16) _Float16 sQ [kNP * 128]; // q * scale, row-major
  __shared__ __align__(16) _Float16 sK [kNP * 128]; // k, row-major (== col-major B for q@k^T)
  __shared__ __align__(16) _Float16 sSP[kNP * 128]; // raw scores, then softmax probs
  __shared__ __align__(16) _Float16 sVT[128 * 128]; // v transposed [d][token], token-padded w/ 0
  __shared__ int sRow[kNP];                         // global row offset per token

  const int tid  = threadIdx.x;
  const int wv   = tid >> 5;
  const int lane = tid & 31;
  const int l16  = lane & 15;
  const int lhi  = lane >> 4;

  const int wid = blockIdx.x;   // 0..4095
  const int b   = wid >> 9;
  const int r   = wid & 511;    // window-in-batch == mask index
  const int wd  = r >> 6;
  const int wh  = (r >> 3) & 7;
  const int ww  = r & 7;

  // ---- zero K-dim padding regions (vT fully; LN rows 98..111) ----
  {
    uint4 z = make_uint4(0u, 0u, 0u, 0u);
    uint4* zv = (uint4*)sVT;
    for (int i = tid; i < (128 * 128 * 2) / 16; i += 256) zv[i] = z;
    uint4* zx = (uint4*)(sXC + kN * 128);
    for (int i = tid; i < ((kNP - kN) * 128 * 2) / 16; i += 256) zx[i] = z;
  }

  // ---- phase 1: gather shifted window tokens + LayerNorm -> f16 ----
  {
    const float4 g4 = ((const float4*)gamma)[lane];
    const float4 b4 = ((const float4*)beta)[lane];
    for (int t = wv; t < kN; t += 8) {
      int ld = t / 49, rm = t % 49, lh = rm / 7, lw = rm % 7;
      int d = wd * 2 + ld, h = wh * 7 + lh, w = ww * 7 + lw;
      int ds = (d + 1) & 15;                 // roll(-1) source == roll(+1) dest
      int hs = h + 3; if (hs >= 56) hs -= 56;
      int ws = w + 3; if (ws >= 56) ws -= 56;
      int gofs = ((b * 16 + ds) * 56 + hs) * 56 + ws;
      if (lane == 0) sRow[t] = gofs;
      float4 xv = ((const float4*)(x + (size_t)gofs * 128))[lane];
      float s  = xv.x + xv.y + xv.z + xv.w;
      float ss = xv.x * xv.x + xv.y * xv.y + xv.z * xv.z + xv.w * xv.w;
      s  = wave_sum(s);
      ss = wave_sum(ss);
      float mu   = s * (1.0f / 128.0f);
      float var  = ss * (1.0f / 128.0f) - mu * mu;
      float rstd = rsqrtf(var + 1e-5f);
      _Float16* dst = &sXC[t * 128 + lane * 4];
      dst[0] = (_Float16)((xv.x - mu) * rstd * g4.x + b4.x);
      dst[1] = (_Float16)((xv.y - mu) * rstd * g4.y + b4.y);
      dst[2] = (_Float16)((xv.z - mu) * rstd * g4.z + b4.z);
      dst[3] = (_Float16)((xv.w - mu) * rstd * g4.w + b4.w);
    }
  }
  __syncthreads();

  // ---- phase 2: qkv = LN(x) @ wqkv^T + bqkv ; scatter into q/k/vT ----
  for (int nti = 0; nti < 3; ++nti) {
    const int nt   = wv * 3 + nti;        // 0..23 output column tiles (384 cols)
    const int ncol = nt * 16 + l16;
    v16h Bf0 = load_b_w32(wqkv, ncol, 0,  lane);
    v16h Bf1 = load_b_w32(wqkv, ncol, 32, lane);
    v16h Bf2 = load_b_w32(wqkv, ncol, 64, lane);
    v16h Bf3 = load_b_w32(wqkv, ncol, 96, lane);
    const float bias = bqkv[ncol];
    for (int mt = 0; mt < 7; ++mt) {
      const int row = mt * 16 + l16;
      // preload all A fragments, then issue WMMAs back-to-back (1 dscnt wait)
      v16h A0 = load_a(sXC, row, 0,  lane);
      v16h A1 = load_a(sXC, row, 32, lane);
      v16h A2 = load_a(sXC, row, 64, lane);
      v16h A3 = load_a(sXC, row, 96, lane);
      v8f acc = vfill(bias);
      acc = wmma_f16(A0, Bf0, acc);
      acc = wmma_f16(A1, Bf1, acc);
      acc = wmma_f16(A2, Bf2, acc);
      acc = wmma_f16(A3, Bf3, acc);
      // wave-uniform routing on nt (16-col tiles never straddle 128/256)
      if (nt < 8) {
#pragma unroll
        for (int i = 0; i < 8; ++i) {
          int m = mt * 16 + i + (lhi << 3);
          sQ[m * 128 + ncol] = (_Float16)(acc[i] * kSCALE);
        }
      } else if (nt < 16) {
        const int kcol = ncol - 128;
#pragma unroll
        for (int i = 0; i < 8; ++i) {
          int m = mt * 16 + i + (lhi << 3);
          sK[m * 128 + kcol] = (_Float16)acc[i];
        }
      } else {
        const int dcol = ncol - 256;
        if (mt < 6) {                      // rows all < 96 < kN
#pragma unroll
          for (int i = 0; i < 8; ++i) {
            int m = mt * 16 + i + (lhi << 3);
            sVT[dcol * 128 + m] = (_Float16)acc[i];
          }
        } else if (lhi == 0) {             // mt==6: only rows 96,97 valid
          sVT[dcol * 128 + 96] = (_Float16)acc[0];
          sVT[dcol * 128 + 97] = (_Float16)acc[1];
        }
      }
    }
  }
  __syncthreads();

  // ---- phases 3-4: attention, one head at a time ----
  for (int hd = 0; hd < kNH; ++hd) {
    // scores = q@k^T + rel_bias + mask
    for (int tile = wv; tile < 49; tile += 8) {
      int mt = tile / 7, nt = tile % 7;
      v16h Af  = load_a(sQ, mt * 16 + l16, hd * 32, lane);
      v16h Bfr = load_b(sK, nt * 16 + l16, hd * 32, lane);
      v8f acc = wmma_f16(Af, Bfr, vfill(0.0f));
      const int n  = nt * 16 + l16;
      const int nc = (n < kN - 1) ? n : (kN - 1);      // clamp for safe gathers
      const bool nok = (n < kN);
#pragma unroll
      for (int i = 0; i < 8; ++i) {
        int m  = mt * 16 + i + (lhi << 3);
        int mc = (m < kN - 1) ? m : (kN - 1);
        // unconditional clamped gathers -> no exec-mask branching
        int   idx = rel[mc * kN + nc];
        float bm  = btab[idx * kNH + hd] + mask[((size_t)r * kN + mc) * kN + nc];
        float sv  = (nok && m < kN) ? (acc[i] + bm) : 0.0f;
        sSP[m * 128 + n] = (_Float16)sv;
      }
    }
    __syncthreads();
    // softmax over columns 0..97; write probs in place (cols 98..127 -> 0)
    for (int row = wv; row < kN; row += 8) {
      _Float16* sr = &sSP[row * 128];
      float s0 = (float)sr[lane];
      float s1 = (float)sr[lane + 32];
      float s2 = (float)sr[lane + 64];
      float s3 = (lane < 2) ? (float)sr[lane + 96] : -1e30f;
      float mx = wave_max(fmaxf(fmaxf(s0, s1), fmaxf(s2, s3)));
      float e0 = __expf(s0 - mx);
      float e1 = __expf(s1 - mx);
      float e2 = __expf(s2 - mx);
      float e3 = (lane < 2) ? __expf(s3 - mx) : 0.0f;
      float inv = 1.0f / wave_sum(e0 + e1 + e2 + e3);
      sr[lane]      = (_Float16)(e0 * inv);
      sr[lane + 32] = (_Float16)(e1 * inv);
      sr[lane + 64] = (_Float16)(e2 * inv);
      sr[lane + 96] = (_Float16)(e3 * inv);   // zeros for cols >= 98
    }
    __syncthreads();
    // ctx[:, head-slice] = P @ v   (K = 128 padded tokens, zeros beyond 97)
    for (int tile = wv; tile < 14; tile += 8) {
      int mt = tile >> 1, ntd = tile & 1;
      const int row = mt * 16 + l16;
      const int col = hd * 32 + ntd * 16 + l16;
      v16h A0 = load_a(sSP, row, 0,  lane);
      v16h A1 = load_a(sSP, row, 32, lane);
      v16h A2 = load_a(sSP, row, 64, lane);
      v16h A3 = load_a(sSP, row, 96, lane);
      v16h B0 = load_b(sVT, col, 0,  lane);
      v16h B1 = load_b(sVT, col, 32, lane);
      v16h B2 = load_b(sVT, col, 64, lane);
      v16h B3 = load_b(sVT, col, 96, lane);
      v8f acc = vfill(0.0f);
      acc = wmma_f16(A0, B0, acc);
      acc = wmma_f16(A1, B1, acc);
      acc = wmma_f16(A2, B2, acc);
      acc = wmma_f16(A3, B3, acc);
#pragma unroll
      for (int i = 0; i < 8; ++i) {
        int m = mt * 16 + i + (lhi << 3);
        sXC[m * 128 + col] = (_Float16)acc[i];
      }
    }
    __syncthreads();
  }

  // ---- phase 5: proj + scatter back (window reverse + reverse shift) ----
  {
    const int ncol = wv * 16 + l16;        // 8 waves x 16 = 128 output cols
    v16h Bf0 = load_b_w32(wproj, ncol, 0,  lane);
    v16h Bf1 = load_b_w32(wproj, ncol, 32, lane);
    v16h Bf2 = load_b_w32(wproj, ncol, 64, lane);
    v16h Bf3 = load_b_w32(wproj, ncol, 96, lane);
    const float bias = bproj[ncol];
    for (int mt = 0; mt < 7; ++mt) {
      const int row = mt * 16 + l16;
      v16h A0 = load_a(sXC, row, 0,  lane);
      v16h A1 = load_a(sXC, row, 32, lane);
      v16h A2 = load_a(sXC, row, 64, lane);
      v16h A3 = load_a(sXC, row, 96, lane);
      v8f acc = vfill(bias);
      acc = wmma_f16(A0, Bf0, acc);
      acc = wmma_f16(A1, Bf1, acc);
      acc = wmma_f16(A2, Bf2, acc);
      acc = wmma_f16(A3, Bf3, acc);
      if (mt < 6) {                        // all 8 rows valid
#pragma unroll
        for (int i = 0; i < 8; ++i) {
          int m = mt * 16 + i + (lhi << 3);
          out[(size_t)sRow[m] * 128 + ncol] = acc[i];
        }
      } else if (lhi == 0) {               // mt==6: rows 96,97 only
        out[(size_t)sRow[96] * 128 + ncol] = acc[0];
        out[(size_t)sRow[97] * 128 + ncol] = acc[1];
      }
    }
  }
}

extern "C" void kernel_launch(void* const* d_in, const int* in_sizes, int n_in,
                              void* d_out, int out_size, void* d_ws, size_t ws_size,
                              hipStream_t stream) {
  (void)in_sizes; (void)n_in; (void)d_ws; (void)ws_size; (void)out_size;
  const float* x     = (const float*)d_in[0];
  const float* mask  = (const float*)d_in[1];
  const float* gamma = (const float*)d_in[2];
  const float* beta  = (const float*)d_in[3];
  const float* wqkv  = (const float*)d_in[4];
  const float* bqkv  = (const float*)d_in[5];
  const float* wproj = (const float*)d_in[6];
  const float* bproj = (const float*)d_in[7];
  const float* btab  = (const float*)d_in[8];
  const int*   rel   = (const int*)d_in[9];
  float* outp = (float*)d_out;
  dim3 grid(8 * 512);   // one workgroup per window
  dim3 block(256);      // 8 wave32
  swin3d_window_kernel<<<grid, block, 0, stream>>>(
      x, mask, gamma, beta, wqkv, bqkv, wproj, bproj, btab, rel, outp);
}
